// SelfAttention_76768245449012
// MI455X (gfx1250) — matrix operand order repaired and checked by reading, not compiled, not run on hardware
//
#include <hip/hip_runtime.h>
#include <hip/hip_bf16.h>

// ---------------------------------------------------------------------------
// Self-attention (B=4, C=256, H=W=64 -> N=4096, E=64) for gfx1250 (MI455X).
// Flash-attention formulation: never materialize the [B,N,N] score matrix.
// All GEMMs via v_wmma_f32_16x16x32_bf16 (wave32 WMMA).
// Weights pre-converted to bf16 once so hot loops carry no cvt chains.
// ---------------------------------------------------------------------------

typedef __attribute__((ext_vector_type(16))) __bf16 v16bf;
typedef __attribute__((ext_vector_type(8)))  __bf16 v8bf;
typedef __attribute__((ext_vector_type(8)))  float  v8f;

#define NB   4
#define CC   256
#define NN   4096
#define EE   64
#define WEC  (EE * CC)          // 16384 elements per weight matrix

#define WMMA_BF16(A, Bm, Cm) \
    __builtin_amdgcn_wmma_f32_16x16x32_bf16(false, (A), false, (Bm), (short)0, (Cm), false, false)

static __device__ __forceinline__ v8f vzero8() {
    v8f z;
#pragma unroll
    for (int i = 0; i < 8; ++i) z[i] = 0.0f;
    return z;
}

// 16-bit A-operand (16x32, row-major source, contiguous k):
// lane L holds row (L&15); elements 0..15 = k 2*kb..2*kb+15,
// kb = (L>>4)*8.  rowPtr points at the row's k0.
static __device__ __forceinline__ v16bf mkA(const __bf16* rowPtr, int kb) {
    v8bf lo = *(const v8bf*)(rowPtr + 2 * kb);
    v8bf hi = *(const v8bf*)(rowPtr + 2 * kb + 8);
    v16bf a;
#pragma unroll
    for (int t = 0; t < 8; ++t) { a[t] = lo[t]; a[8 + t] = hi[t]; }
    return a;
}

// ---------------------------------------------------------------------------
// Kernel 0: one-shot bf16 conversion of the four weight matrices.
// ---------------------------------------------------------------------------
__global__ void __launch_bounds__(256) cvt_weights(
    const float* __restrict__ Wk, const float* __restrict__ Wq,
    const float* __restrict__ Wv, const float* __restrict__ Wo,
    __bf16* __restrict__ Wkb, __bf16* __restrict__ Wqb,
    __bf16* __restrict__ Wvb, __bf16* __restrict__ Wob)
{
    int i = blockIdx.x * 256 + threadIdx.x;
    if (i < WEC) {
        Wkb[i] = (__bf16)Wk[i];
        Wqb[i] = (__bf16)Wq[i];
        Wvb[i] = (__bf16)Wv[i];
        Wob[i] = (__bf16)Wo[i];
    }
}

// ---------------------------------------------------------------------------
// Kernel 1: K/Q/V projections (1x1 convs) with WMMA.
//   K, Q stored [B][N][E] bf16 (row = position, e contiguous)
//   V  stored transposed [B][E][N] bf16 so the PV GEMM's A-operand is contiguous
// ---------------------------------------------------------------------------
__global__ void __launch_bounds__(256) proj_kernel(
    const float* __restrict__ x,
    const __bf16* __restrict__ Wkb, const float* __restrict__ bk,
    const __bf16* __restrict__ Wqb, const float* __restrict__ bq,
    const __bf16* __restrict__ Wvb, const float* __restrict__ bv,
    __bf16* __restrict__ Kw, __bf16* __restrict__ Qw, __bf16* __restrict__ Vt)
{
    const int b    = blockIdx.x >> 5;           // 32 position-tiles per batch
    const int n0   = (blockIdx.x & 31) * 128;   // 128 positions per block
    const int tid  = threadIdx.x;
    const int lane = tid & 31, wave = tid >> 5;
    const int lhalf = lane >> 4, l15 = lane & 15;
    const int kb   = lhalf * 8;
    const int kb16 = lhalf * 16;

    // x chunk staged transposed: [n_local][c_local] so A-loads are contiguous
    __shared__ __align__(32) __bf16 xT[128 * 32];

    v8f acc[12];                                 // {K,Q,V} x 4 e-tiles
#pragma unroll
    for (int i = 0; i < 12; ++i) acc[i] = vzero8();

    const __bf16* Wmat[3] = {Wkb, Wqb, Wvb};

    for (int c0 = 0; c0 < CC; c0 += 32) {
        for (int idx = tid; idx < 32 * 128; idx += 256) {
            int cc = idx >> 7;                   // 0..31  (c within chunk)
            int nn = idx & 127;                  // 0..127 (position in tile)
            float v = x[((size_t)b * CC + (c0 + cc)) * NN + (n0 + nn)];
            xT[nn * 32 + cc] = (__bf16)v;
        }
        __syncthreads();

        // A: this wave's 16 positions x 32 channels (from LDS, contiguous)
        const __bf16* arow = &xT[(16 * wave + l15) * 32];
        v16bf A = mkA(arow, kb);

#pragma unroll
        for (int wsel = 0; wsel < 3; ++wsel) {
            const __bf16* W = Wmat[wsel];
#pragma unroll
            for (int et = 0; et < 4; ++et) {
                // B[k=c][n=e] = W[e][c0+k]: lane holds col e, 16 contiguous c
                v16bf Bm = *(const v16bf*)(W + (size_t)(et * 16 + l15) * CC + c0 + kb16);
                acc[wsel * 4 + et] = WMMA_BF16(A, Bm, acc[wsel * 4 + et]);
            }
        }
        __syncthreads();
    }

    // D layout: col e = 16*et + l15, rows = position 16*wave + 8*lhalf + r
#pragma unroll
    for (int et = 0; et < 4; ++et) {
        int e = et * 16 + l15;
        float bbk = bk[e], bbq = bq[e], bbv = bv[e];
#pragma unroll
        for (int r = 0; r < 8; ++r) {
            int n = n0 + 16 * wave + 8 * lhalf + r;
            Kw[((size_t)b * NN + n) * EE + e] = (__bf16)(acc[0 + et][r] + bbk);
            Qw[((size_t)b * NN + n) * EE + e] = (__bf16)(acc[4 + et][r] + bbq);
            Vt[((size_t)b * EE + e) * NN + n] = (__bf16)(acc[8 + et][r] + bbv);
        }
    }
}

// ---------------------------------------------------------------------------
// Kernel 2: flash attention (softmax over keys i, per query column j) fused
// with the Wo output projection, bias, gamma and residual add.
// Each block: 128 queries; each wave: 16 queries, full online softmax state.
// 64 keys per iteration: 16 WMMAs per softmax pass for a good VALU:XDL ratio.
// ---------------------------------------------------------------------------
__global__ void __launch_bounds__(256) attn_kernel(
    const float* __restrict__ x,
    const __bf16* __restrict__ Wob, const float* __restrict__ bo,
    const float* __restrict__ gammap,
    const __bf16* __restrict__ Kw, const __bf16* __restrict__ Qw,
    const __bf16* __restrict__ Vt,
    float* __restrict__ out)
{
    const int b    = blockIdx.x >> 5;
    const int n0   = (blockIdx.x & 31) * 128;
    const int tid  = threadIdx.x;
    const int lane = tid & 31, wave = tid >> 5;
    const int lhalf = lane >> 4, l15 = lane & 15;
    const int kb   = lhalf * 8;
    const int kb16 = lhalf * 16;

    __shared__ __align__(32) __bf16 Vtile[128 * EE];   // [j_local][e]

    // Resident Q B-operands for this wave's 16 queries (k = e, 2 chunks of 32)
    const __bf16* qrow = Qw + ((size_t)b * NN + (n0 + 16 * wave + l15)) * EE;
    v16bf QB0 = *(const v16bf*)(qrow + kb16);          // e 0..31
    v16bf QB1 = *(const v16bf*)(qrow + 32 + kb16);     // e 32..63

    v8f acc[4];                                        // v[e-tile][j]
#pragma unroll
    for (int i = 0; i < 4; ++i) acc[i] = vzero8();
    float mrun = -INFINITY, lrun = 0.0f;

    const __bf16* kbase = Kw + (size_t)b * NN * EE;
    const __bf16* vbase = Vt + (size_t)b * EE * NN;

    for (int i0 = 0; i0 < NN; i0 += 64) {              // 64 keys / iteration
        // --- s tiles: D[m=i][n=j] = sum_e K[i][e] Q[j][e], 4 i-subtiles ----
        v8f s[4];
#pragma unroll
        for (int t = 0; t < 4; ++t) {
            const __bf16* krow = kbase + (size_t)(i0 + 16 * t + l15) * EE;
            v8f st = vzero8();
            st = WMMA_BF16(mkA(krow,      kb), QB0, st);
            st = WMMA_BF16(mkA(krow + 32, kb), QB1, st);
            s[t] = st;
        }
        // prefetch next key chunk toward the WGP while softmax runs
        if (i0 + 64 < NN) {
            __builtin_prefetch(kbase + (size_t)(i0 + 64 + l15) * EE, 0, 0);
            __builtin_prefetch(vbase + (size_t)l15 * NN + i0 + 64, 0, 0);
        }

        // --- online softmax over this chunk's 64 keys (per column j) -------
        // Column j lives in lanes j and j+16 -> one shfl_xor(16) reduction.
        float cmax = -INFINITY;
#pragma unroll
        for (int t = 0; t < 4; ++t)
#pragma unroll
            for (int r = 0; r < 8; ++r) cmax = fmaxf(cmax, s[t][r]);
        cmax = fmaxf(cmax, __shfl_xor(cmax, 16, 32));
        float mnew  = fmaxf(mrun, cmax);
        float scale = __expf(mrun - mnew);
        float psum = 0.0f;
#pragma unroll
        for (int t = 0; t < 4; ++t)
#pragma unroll
            for (int r = 0; r < 8; ++r) {
                s[t][r] = __expf(s[t][r] - mnew);      // p in place of s
                psum += s[t][r];
            }
        psum += __shfl_xor(psum, 16, 32);
        lrun = lrun * scale + psum;
        mrun = mnew;

        // --- repack p (D-layout) into two 32x16 B-operands -----------------
        // BP[c] covers keys 32c..32c+31 from tile pair (2c, 2c+1):
        // lane<16 needs rows 0..15 of tile 2c; lane>=16 rows 0..15 of 2c+1;
        // exchange the missing 8 rows with lane^16.
        v16bf BP[2];
#pragma unroll
        for (int cpair = 0; cpair < 2; ++cpair) {
            const v8f& pa = s[2 * cpair];              // keys +0..15
            const v8f& pb = s[2 * cpair + 1];          // keys +16..31
            float recv[8];
#pragma unroll
            for (int r = 0; r < 8; ++r) {
                float sendv = lhalf ? pa[r] : pb[r];
                recv[r] = __shfl_xor(sendv, 16, 32);
            }
            v16bf bp;
#pragma unroll
            for (int r = 0; r < 8; ++r) {
                bp[r]     = (__bf16)(lhalf ? recv[r] : pa[r]);
                bp[8 + r] = (__bf16)(lhalf ? pb[r]   : recv[r]);
            }
            BP[cpair] = bp;
        }

        // --- v accumulate: D[m=e][n=j] += Vt[e][i] * p[i][j] ---------------
#pragma unroll
        for (int et = 0; et < 4; ++et) {
            const __bf16* vrow = vbase + (size_t)(et * 16 + l15) * NN + i0;
#pragma unroll
            for (int r = 0; r < 8; ++r) acc[et][r] *= scale;
            acc[et] = WMMA_BF16(mkA(vrow,      kb), BP[0], acc[et]);
            acc[et] = WMMA_BF16(mkA(vrow + 32, kb), BP[1], acc[et]);
        }
    }

    // normalize and stage v tile in LDS as [j][e] (contiguous e for B-loads)
    float rinv = 1.0f / lrun;
    const int jl = 16 * wave + l15;
#pragma unroll
    for (int et = 0; et < 4; ++et) {
#pragma unroll
        for (int r = 0; r < 8; ++r) {
            int e = et * 16 + 8 * lhalf + r;
            Vtile[jl * EE + e] = (__bf16)(acc[et][r] * rinv);
        }
    }
    __syncthreads();

    // ---- fused output projection: o = Wo * v + bo ; y = gamma*o + x -------
    const float gamma = *gammap;
#pragma unroll
    for (int cti = 0; cti < 2; ++cti) {
        int ct = 2 * wave + cti;                      // c-tile 0..15
        const __bf16* worow = Wob + (size_t)(ct * 16 + l15) * EE;
        v16bf WA0 = mkA(worow,      kb);              // A: [m=c][k=e 0..31]
        v16bf WA1 = mkA(worow + 32, kb);              //            e 32..63
#pragma unroll
        for (int jt = 0; jt < 8; ++jt) {
            const __bf16* vrow = &Vtile[(16 * jt + l15) * EE];
            v16bf B0 = *(const v16bf*)(vrow + kb16);
            v16bf B1 = *(const v16bf*)(vrow + 32 + kb16);
            v8f d = vzero8();
            d = WMMA_BF16(WA0, B0, d);
            d = WMMA_BF16(WA1, B1, d);
            int n = n0 + 16 * jt + l15;
#pragma unroll
            for (int r = 0; r < 8; ++r) {
                int c = ct * 16 + 8 * lhalf + r;
                size_t idx = ((size_t)b * CC + c) * NN + n;
                out[idx] = gamma * (d[r] + bo[c]) + x[idx];
            }
        }
    }
}

// ---------------------------------------------------------------------------
extern "C" void kernel_launch(void* const* d_in, const int* in_sizes, int n_in,
                              void* d_out, int out_size, void* d_ws, size_t ws_size,
                              hipStream_t stream) {
    const float* x     = (const float*)d_in[0];
    const float* Wk    = (const float*)d_in[1];
    const float* bk    = (const float*)d_in[2];
    const float* Wq    = (const float*)d_in[3];
    const float* bq    = (const float*)d_in[4];
    const float* Wv    = (const float*)d_in[5];
    const float* bv    = (const float*)d_in[6];
    const float* Wo    = (const float*)d_in[7];
    const float* bo    = (const float*)d_in[8];
    const float* gamma = (const float*)d_in[9];
    float* out = (float*)d_out;

    // workspace: K,Q [B][N][E] + V transposed [B][E][N] bf16 (6 MB)
    //          + bf16 copies of Wk,Wq,Wv,Wo (4 x 32 KB)
    __bf16* Kw  = (__bf16*)d_ws;
    __bf16* Qw  = Kw + (size_t)NB * NN * EE;
    __bf16* Vt  = Qw + (size_t)NB * NN * EE;
    __bf16* Wkb = Vt + (size_t)NB * NN * EE;
    __bf16* Wqb = Wkb + WEC;
    __bf16* Wvb = Wqb + WEC;
    __bf16* Wob = Wvb + WEC;

    dim3 grid(NB * (NN / 128));   // 128 blocks
    dim3 block(256);              // 8 waves (wave32)
    cvt_weights<<<dim3(WEC / 256), block, 0, stream>>>(Wk, Wq, Wv, Wo, Wkb, Wqb, Wvb, Wob);
    proj_kernel<<<grid, block, 0, stream>>>(x, Wkb, bk, Wqb, bq, Wvb, bv, Kw, Qw, Vt);
    attn_kernel<<<grid, block, 0, stream>>>(x, Wob, bo, gamma, Kw, Qw, Vt, out);
}
